// W_SimVQ_decompose_19765439496199
// MI455X (gfx1250) — matrix operand order repaired and checked by reading, not compiled, not run on hardware
//
#include <hip/hip_runtime.h>
#include <hip/hip_bf16.h>
#include <stdint.h>

#define DIM    1024
#define NEMB   8192
#define NROWS  32768                      // 32 * 1024 tokens
#define ZELEMS ((size_t)NROWS * DIM)      // z_out element count

typedef __attribute__((ext_vector_type(16))) __bf16 v16bf;
typedef __attribute__((ext_vector_type(8)))  float  v8f;
typedef __attribute__((ext_vector_type(4)))  int    v4i;

#define AS1 __attribute__((address_space(1)))
#define AS3 __attribute__((address_space(3)))

// ---- CDNA5 async global->LDS path (guarded; falls back to sync staging) ----
#if defined(__has_builtin)
#if __has_builtin(__builtin_amdgcn_global_load_async_to_lds_b128)
#define HAVE_ASYNC_LDS 1
#endif
#endif
#ifndef HAVE_ASYNC_LDS
#define HAVE_ASYNC_LDS 0
#endif

#if HAVE_ASYNC_LDS
#if defined(__has_builtin) && __has_builtin(__builtin_amdgcn_s_wait_asynccnt)
#define WAIT_ASYNC() __builtin_amdgcn_s_wait_asynccnt(0)
#else
#define WAIT_ASYNC() asm volatile("s_wait_asynccnt 0x0" ::: "memory")
#endif
#else
#define WAIT_ASYNC() do {} while (0)
#endif

// copy one 16B chunk global -> LDS (async on CDNA5, tracked by ASYNCcnt)
__device__ __forceinline__ void cp16(unsigned short* lds_dst, const unsigned short* gsrc) {
#if HAVE_ASYNC_LDS
  __builtin_amdgcn_global_load_async_to_lds_b128(
      (AS1 v4i*)const_cast<unsigned short*>(gsrc), (AS3 v4i*)lds_dst, 0, 0);
#else
  *(uint4*)lds_dst = *(const uint4*)gsrc;
#endif
}

// ---------- scalar helpers ----------

// fp32 -> bf16, round-to-nearest-even
__device__ __forceinline__ unsigned short f2bf(float f) {
  unsigned int u = __float_as_uint(f);
  unsigned int r = u + 0x7FFFu + ((u >> 16) & 1u);
  return (unsigned short)(r >> 16);
}
__device__ __forceinline__ float bf2f(unsigned short h) {
  return __uint_as_float((unsigned int)h << 16);
}
__device__ __forceinline__ unsigned int pk2(float a, float b) {
  return (unsigned int)f2bf(a) | ((unsigned int)f2bf(b) << 16);
}
// split fp32 pair into (hi, lo) bf16 packed pairs: a = hi + lo (+eps)
__device__ __forceinline__ void hilopk(float a, float b,
                                       unsigned int& hp, unsigned int& lp) {
  unsigned short ha = f2bf(a), hb = f2bf(b);
  float la = a - bf2f(ha);
  float lb = b - bf2f(hb);
  hp = (unsigned int)ha | ((unsigned int)hb << 16);
  lp = (unsigned int)f2bf(la) | ((unsigned int)f2bf(lb) << 16);
}

// A operand (16x32 bf16): lanes 0-15 hold K {0..15},
// lanes 16-31 hold K {16..31} for row M = lane&15.
__device__ __forceinline__ v16bf ld_a(const unsigned short* row, int ka) {
  union { uint4 q[2]; v16bf v; } u;
  u.q[0] = *(const uint4*)(row + ka);        // 8 bf16
  u.q[1] = *(const uint4*)(row + ka + 8);    // 8 bf16
  return u.v;
}
// B operand (32x16 bf16): lane L holds contiguous K run of 16 for N = L&15.
__device__ __forceinline__ v16bf ld_b(const unsigned short* p) {
  union { uint4 q[2]; v16bf v; } u;
  u.q[0] = *(const uint4*)(p);
  u.q[1] = *(const uint4*)(p + 8);
  return u.v;
}

// order-preserving float -> uint (for packed atomicMin argmin keys)
__device__ __forceinline__ unsigned int forder(float f) {
  unsigned int u = __float_as_uint(f);
  return (u & 0x80000000u) ? ~u : (u ^ 0x80000000u);
}

// ---------- kernel: distance GEMM (bf16 WMMA, async double-buffered LDS) + fused argmin ----
// score[n][k] = ||e_k||^2 - 2 * x_n . e_k  (||x||^2 dropped: constant per row)
// grid (K/128, NROWS/128), 256 threads (8 waves). Block tile 128 rows x 128 codes.
// Wave tile 32x64 (2 M-subtiles x 4 N-subtiles, 8 v8f accumulators).
// BK = 32 per stage, 32 stages, double-buffered via ASYNCcnt.
__global__ __launch_bounds__(256) void dist_argmin_kernel(
    const unsigned short* __restrict__ xb, const unsigned short* __restrict__ Cb,
    const float* __restrict__ norms, unsigned long long* __restrict__ keys) {
  __shared__ unsigned short xs[2][128][40];   // 20 KB
  __shared__ unsigned short cs[2][128][40];   // 20 KB
  const int t = threadIdx.x, lane = t & 31, wave = t >> 5;
  const int wm = wave & 3, wn = wave >> 2;   // 4 M strips x 2 N strips
  const int rbase = blockIdx.y * 128;   // token rows
  const int cbase = blockIdx.x * 128;   // codes
  v8f acc[2][4] = {};

  auto stage = [&](int s, int buf) {
    const int d0 = s * 32;
    // x tile 128x32 bf16 = 512 x 16B chunks, 2 per thread
#pragma unroll
    for (int c = 0; c < 2; ++c) {
      const int chunk = t + c * 256;
      const int r = chunk >> 2, col = (chunk & 3) * 8;
      cp16(&xs[buf][r][col], xb + (size_t)(rbase + r) * DIM + d0 + col);
    }
    // codebook tile 128x32 bf16 = 512 x 16B chunks, 2 per thread
#pragma unroll
    for (int c = 0; c < 2; ++c) {
      const int chunk = t + c * 256;
      const int r = chunk >> 2, col = (chunk & 3) * 8;
      cp16(&cs[buf][r][col], Cb + (size_t)(cbase + r) * DIM + d0 + col);
    }
  };

  stage(0, 0);
  WAIT_ASYNC();
  __syncthreads();

  const int ka = (lane >> 4) * 16;
  const int kb = (lane >> 4) * 16;
  const int NSTAGE = DIM / 32;

  for (int s = 0; s < NSTAGE; ++s) {
    const int cur = s & 1;
    if (s + 1 < NSTAGE) stage(s + 1, cur ^ 1);  // next tiles in flight while we compute
    v16bf a[2];
#pragma unroll
    for (int mt = 0; mt < 2; ++mt)
      a[mt] = ld_a(&xs[cur][wm * 32 + mt * 16 + (lane & 15)][0], ka);
#pragma unroll
    for (int i = 0; i < 4; ++i) {
      const int brow = wn * 64 + i * 16 + (lane & 15);
      v16bf bm = ld_b(&cs[cur][brow][kb]);
#pragma unroll
      for (int mt = 0; mt < 2; ++mt)
        acc[mt][i] = __builtin_amdgcn_wmma_f32_16x16x32_bf16(
            false, a[mt], false, bm, (short)0, acc[mt][i], false, false);
    }
    if (s + 1 < NSTAGE) {
      WAIT_ASYNC();      // LDS writes for stage s+1 landed
      __syncthreads();   // visible to all waves; other buffer reusable
    }
  }

  // per-row argmin over this block's 128 codes, then global combine
  float nv[4];
  int gn[4];
#pragma unroll
  for (int i = 0; i < 4; ++i) {
    gn[i] = cbase + wn * 64 + i * 16 + (lane & 15);
    nv[i] = norms[gn[i]];
  }
#pragma unroll
  for (int mt = 0; mt < 2; ++mt) {
#pragma unroll
    for (int r = 0; r < 8; ++r) {
      float best = 3.4e38f;
      int bidx = 0x7fffffff;
#pragma unroll
      for (int i = 0; i < 4; ++i) {
        const float s = nv[i] - 2.0f * acc[mt][i][r];
        if (s < best) { best = s; bidx = gn[i]; }
      }
      // reduce across the 16 lanes of each half-wave (rows differ between halves)
      for (int off = 1; off < 16; off <<= 1) {
        const float ob = __shfl_xor(best, off, 32);
        const int   oi = __shfl_xor(bidx, off, 32);
        if (ob < best || (ob == best && oi < bidx)) { best = ob; bidx = oi; }
      }
      if ((lane & 15) == 0) {
        const int grow = rbase + wm * 32 + mt * 16 + ((lane >> 4) << 3) + r;
        const unsigned long long key =
            ((unsigned long long)forder(best) << 32) | (unsigned int)bidx;
        atomicMin(&keys[grow], key);  // exact + deterministic; ties -> lowest index
      }
    }
  }
}

// ---------- kernel: codebook projection C = embed @ W^T + b (split-bf16 WMMA ~fp32) ----------
// grid (DIM/128, NEMB/64), 256 threads (8 waves). Block tile 64(k) x 128(j).
__global__ __launch_bounds__(256) void proj_gemm_kernel(
    const float* __restrict__ embed, const float* __restrict__ W,
    const float* __restrict__ bias, float* __restrict__ C) {
  __shared__ unsigned short ehi[64][40], elo[64][40];
  __shared__ unsigned short whi[128][40], wlo[128][40];
  const int t = threadIdx.x, lane = t & 31, wave = t >> 5;
  const int wm = wave & 3, wn = wave >> 2;
  const int kbase = blockIdx.y * 64;     // embedding rows
  const int jbase = blockIdx.x * 128;    // output columns
  v8f acc[4] = {};

  for (int d0 = 0; d0 < DIM; d0 += 32) {
    { // stage embed 64x32 -> hi/lo bf16
      const int r = t >> 2, c0 = (t & 3) * 8;
      const float* src = embed + (size_t)(kbase + r) * DIM + d0 + c0;
      float4 f0 = ((const float4*)src)[0];
      float4 f1 = ((const float4*)src)[1];
      float f[8] = {f0.x, f0.y, f0.z, f0.w, f1.x, f1.y, f1.z, f1.w};
      unsigned int hp[4], lp[4];
#pragma unroll
      for (int i = 0; i < 4; ++i) hilopk(f[2 * i], f[2 * i + 1], hp[i], lp[i]);
      *(uint4*)&ehi[r][c0] = make_uint4(hp[0], hp[1], hp[2], hp[3]);
      *(uint4*)&elo[r][c0] = make_uint4(lp[0], lp[1], lp[2], lp[3]);
      if (d0 + 32 < DIM) __builtin_prefetch(src + 32, 0, 0);  // global_prefetch_b8
    }
    { // stage W 128x32 -> hi/lo bf16  (B[d][j] = W[j][d], contiguous in W rows)
      const int r = t >> 1, c0 = (t & 1) * 16;
      const float* src = W + (size_t)(jbase + r) * DIM + d0 + c0;
#pragma unroll
      for (int h = 0; h < 2; ++h) {
        float4 f0 = ((const float4*)src)[2 * h];
        float4 f1 = ((const float4*)src)[2 * h + 1];
        float f[8] = {f0.x, f0.y, f0.z, f0.w, f1.x, f1.y, f1.z, f1.w};
        unsigned int hp[4], lp[4];
#pragma unroll
        for (int i = 0; i < 4; ++i) hilopk(f[2 * i], f[2 * i + 1], hp[i], lp[i]);
        *(uint4*)&whi[r][c0 + 8 * h] = make_uint4(hp[0], hp[1], hp[2], hp[3]);
        *(uint4*)&wlo[r][c0 + 8 * h] = make_uint4(lp[0], lp[1], lp[2], lp[3]);
      }
      if (d0 + 32 < DIM) __builtin_prefetch(src + 32, 0, 0);
    }
    __syncthreads();
    const int arow = wm * 16 + (lane & 15);
    const int ka = (lane >> 4) * 16;
    const int kb = (lane >> 4) * 16;
    v16bf ahi = ld_a(&ehi[arow][0], ka);
    v16bf alo = ld_a(&elo[arow][0], ka);
#pragma unroll
    for (int i = 0; i < 4; ++i) {
      const int brow = wn * 64 + i * 16 + (lane & 15);
      v16bf bhi = ld_b(&whi[brow][kb]);
      v16bf blo = ld_b(&wlo[brow][kb]);
      acc[i] = __builtin_amdgcn_wmma_f32_16x16x32_bf16(false, ahi, false, bhi,
                                                       (short)0, acc[i], false, false);
      acc[i] = __builtin_amdgcn_wmma_f32_16x16x32_bf16(false, alo, false, bhi,
                                                       (short)0, acc[i], false, false);
      acc[i] = __builtin_amdgcn_wmma_f32_16x16x32_bf16(false, ahi, false, blo,
                                                       (short)0, acc[i], false, false);
    }
    __syncthreads();
  }
  // epilogue: add bias, store fp32 codebook. C/D layout: VGPR r -> M = r + 8*(lane>=16), N = lane&15.
  const int mrow = wm * 16 + ((lane >> 4) << 3);
#pragma unroll
  for (int i = 0; i < 4; ++i) {
    const int gn = jbase + wn * 64 + i * 16 + (lane & 15);
    const float bj = bias[gn];
#pragma unroll
    for (int r = 0; r < 8; ++r) {
      const int gm = kbase + mrow + r;
      C[(size_t)gm * DIM + gn] = acc[i][r] + bj;
    }
  }
}

// ---------- kernel: row norms ||e_k||^2 + bf16 copy of codebook ----------
__global__ __launch_bounds__(256) void norms_cvt_kernel(
    const float* __restrict__ C, unsigned short* __restrict__ Cb,
    float* __restrict__ norms) {
  __shared__ float red[8];
  const int row = blockIdx.x, t = threadIdx.x;
  const float* cr = C + (size_t)row * DIM;
  unsigned short* br = Cb + (size_t)row * DIM;
  const int j = t * 4;  // 256 threads * 4 = 1024 = DIM, single pass
  float4 v = *(const float4*)(cr + j);
  float s = v.x * v.x + v.y * v.y + v.z * v.z + v.w * v.w;
  uint2 p;
  p.x = pk2(v.x, v.y);
  p.y = pk2(v.z, v.w);
  *(uint2*)(br + j) = p;
#pragma unroll
  for (int off = 16; off; off >>= 1) s += __shfl_down(s, off, 32);
  if ((t & 31) == 0) red[t >> 5] = s;
  __syncthreads();
  if (t == 0) {
    float tot = 0.f;
#pragma unroll
    for (int i = 0; i < 8; ++i) tot += red[i];
    norms[row] = tot;
  }
}

// ---------- kernel: one-shot x fp32 -> bf16 (makes dist staging a pure byte copy) ----------
__global__ __launch_bounds__(256) void x_cvt_kernel(const float* __restrict__ x,
                                                    unsigned short* __restrict__ xb) {
  const size_t i = ((size_t)blockIdx.x * 256 + threadIdx.x) * 8;
  float4 f0 = *(const float4*)(x + i);
  float4 f1 = *(const float4*)(x + i + 4);
  uint4 q;
  q.x = pk2(f0.x, f0.y);
  q.y = pk2(f0.z, f0.w);
  q.z = pk2(f1.x, f1.y);
  q.w = pk2(f1.z, f1.w);
  *(uint4*)(xb + i) = q;
}

// ---------- kernel: reset argmin keys ----------
__global__ void init_kernel(unsigned long long* keys) {
  const int i = blockIdx.x * blockDim.x + threadIdx.x;
  if (i < NROWS) keys[i] = ~0ull;
}

// ---------- kernel: gather z_q, straight-through z_out, per-row diff partials ----------
__global__ __launch_bounds__(256) void gather_diff_kernel(
    const float* __restrict__ x, const float* __restrict__ C,
    const unsigned long long* __restrict__ keys,
    float* __restrict__ z_out, float* __restrict__ idx_out,
    float* __restrict__ partial) {
  __shared__ float red[8];
  const int row = blockIdx.x, t = threadIdx.x;
  const int idx = (int)(keys[row] & 0xFFFFFFFFull);
  if (t == 0) idx_out[row] = (float)idx;
  const float* cr = C + (size_t)idx * DIM;
  const float* xr = x + (size_t)row * DIM;
  float* zr = z_out + (size_t)row * DIM;
  const int j = t * 4;  // single pass: 256 threads * 4 = DIM
  float4 c = *(const float4*)(cr + j);
  float4 xv = *(const float4*)(xr + j);
  float4 d, z;
  d.x = c.x - xv.x; d.y = c.y - xv.y; d.z = c.z - xv.z; d.w = c.w - xv.w;
  z.x = xv.x + d.x; z.y = xv.y + d.y; z.z = xv.z + d.z; z.w = xv.w + d.w;  // x + sg(zq - x)
  *(float4*)(zr + j) = z;
  float s = d.x * d.x + d.y * d.y + d.z * d.z + d.w * d.w;
#pragma unroll
  for (int off = 16; off; off >>= 1) s += __shfl_down(s, off, 32);
  if ((t & 31) == 0) red[t >> 5] = s;
  __syncthreads();
  if (t == 0) {
    float tot = 0.f;
#pragma unroll
    for (int i = 0; i < 8; ++i) tot += red[i];
    partial[row] = tot;
  }
}

// ---------- kernel: deterministic final diff reduction ----------
__global__ __launch_bounds__(256) void finalize_kernel(
    const float* __restrict__ partial, float* __restrict__ out) {
  __shared__ float red[256];
  const int t = threadIdx.x;
  float s = 0.f;
  for (int i = t; i < NROWS; i += 256) s += partial[i];  // fixed order per thread
  red[t] = s;
  __syncthreads();
  for (int off = 128; off; off >>= 1) {
    if (t < off) red[t] += red[t + off];
    __syncthreads();
  }
  // diff = mean((sg(zq)-x)^2) + 0.25*mean((zq-sg(x))^2) = 1.25 * mean((zq-x)^2)
  if (t == 0) out[0] = 1.25f * red[0] / (float)ZELEMS;
}

extern "C" void kernel_launch(void* const* d_in, const int* in_sizes, int n_in,
                              void* d_out, int out_size, void* d_ws, size_t ws_size,
                              hipStream_t stream) {
  const float* x     = (const float*)d_in[0];  // [32,1024,1024]
  const float* embed = (const float*)d_in[1];  // [8192,1024]
  const float* W     = (const float*)d_in[2];  // [1024,1024]
  const float* b     = (const float*)d_in[3];  // [1024]

  float* out      = (float*)d_out;
  float* z_out    = out;                 // [NROWS*DIM]
  float* diff_out = out + ZELEMS;        // [1]
  float* idx_out  = out + ZELEMS + 1;    // [NROWS] (indices as floats)

  // workspace carve (~115 MB)
  char* w = (char*)d_ws;
  float* C = (float*)w;                         w += (size_t)NEMB * DIM * sizeof(float);
  unsigned short* Cb = (unsigned short*)w;      w += (size_t)NEMB * DIM * sizeof(unsigned short);
  unsigned short* xb = (unsigned short*)w;      w += ZELEMS * sizeof(unsigned short);
  float* norms = (float*)w;                     w += (size_t)NEMB * sizeof(float);
  unsigned long long* keys = (unsigned long long*)w; w += (size_t)NROWS * sizeof(unsigned long long);
  float* partial = (float*)w;                   w += (size_t)NROWS * sizeof(float);

  proj_gemm_kernel<<<dim3(DIM / 128, NEMB / 64), 256, 0, stream>>>(embed, W, b, C);
  norms_cvt_kernel<<<NEMB, 256, 0, stream>>>(C, Cb, norms);
  x_cvt_kernel<<<(int)(ZELEMS / (256 * 8)), 256, 0, stream>>>(x, xb);
  init_kernel<<<(NROWS + 255) / 256, 256, 0, stream>>>(keys);
  dist_argmin_kernel<<<dim3(NEMB / 128, NROWS / 128), 256, 0, stream>>>(xb, Cb, norms, keys);
  gather_diff_kernel<<<NROWS, 256, 0, stream>>>(x, C, keys, z_out, idx_out, partial);
  finalize_kernel<<<1, 256, 0, stream>>>(partial, diff_out);
}
